// TemporalCrossTransformer_28071906247071
// MI455X (gfx1250) — matrix operand order repaired and checked by reading, not compiled, not run on hardware
//
#include <hip/hip_runtime.h>
#include <hip/hip_bf16.h>
#include <math.h>

// ---------------- problem constants ----------------
#define SEQ_LEN   8
#define TSS       3
#define D_IN      2048
#define D_OUT     1152
#define WAY       5
#define SHOT      5
#define N_SUPPORT 25
#define N_QUERIES 200
#define N_SAMPLES (N_SUPPORT + N_QUERIES)       // 225
#define T_TUP     56                            // C(8,3)
#define ROWS_ALL  (N_SAMPLES * T_TUP)           // 12600
#define ROWS_S    (N_SUPPORT * T_TUP)           // 1400
#define ROWS_Q    (N_QUERIES * T_TUP)           // 11200
#define K_BIG     (D_IN * TSS)                  // 6144
#define CVT_LD    1408                          // 1400 padded
#define ATT_K     288                           // 280 padded to 32
#define ATT_LDS   296                           // padded row stride (592B, 16B aligned)

typedef __bf16 bf16_t;
typedef __attribute__((ext_vector_type(8)))  __bf16 bf16x8;
typedef __attribute__((ext_vector_type(16))) __bf16 bf16x16;
typedef __attribute__((ext_vector_type(8)))  float  f32x8;
typedef __attribute__((ext_vector_type(4)))  unsigned int u32x4;
typedef __attribute__((ext_vector_type(8)))  int  i32x8;
typedef __attribute__((ext_vector_type(4)))  int  i32x4;

union Frag { bf16x16 v; bf16x8 h[2]; };

__device__ __forceinline__ f32x8 wmma_bf16(bf16x16 a, bf16x16 b, f32x8 c) {
    return __builtin_amdgcn_wmma_f32_16x16x32_bf16(false, a, false, b, (short)0, c, false, false);
}

// ---- CDNA5 async global->LDS copy (per-lane, tracked by ASYNCcnt) ----
__device__ __forceinline__ void async_b128(void* lds_ptr, const void* gptr) {
    unsigned int lds = (unsigned int)(size_t)lds_ptr;  // low 32 bits of generic = LDS offset
    asm volatile("global_load_async_to_lds_b128 %0, %1, off"
                 :: "v"(lds), "v"(gptr) : "memory");
}
__device__ __forceinline__ void wait_async_le4() { asm volatile("s_wait_asynccnt 0x4" ::: "memory"); }
__device__ __forceinline__ void wait_async_0()  { asm volatile("s_wait_asynccnt 0x0" ::: "memory"); }
__device__ __forceinline__ void escape_lds(void* p) { asm volatile("" :: "v"(p) : "memory"); }

// ---- CDNA5 Tensor Data Mover: 2D tile (rem_k x rem_n window), bf16, 64x128 tile,
//      128B lines + 16B LDS pad per line (matches 144B LDS row stride) ----
__device__ __forceinline__ void tdm_load_tile(void* lds_ptr, const void* gptr,
                                              int rem_k, int rem_n, int ld_elems) {
    escape_lds(lds_ptr);
    unsigned long long ga = (unsigned long long)(size_t)gptr;
    u32x4 g0;
    g0[0] = 1u;                                         // count=1, user descriptor
    g0[1] = (unsigned int)(size_t)lds_ptr;              // lds_addr
    g0[2] = (unsigned int)ga;                           // global_addr[31:0]
    g0[3] = (unsigned int)(ga >> 32) | (2u << 30);      // global_addr[56:32] | type=2
    i32x8 g1;
    g1[0] = (1 << 16)            // data_size = 2 bytes
          | (1 << 20)            // pad_enable
          | (4 << 22)            // pad_interval: 32 DWORDs (128B)
          | (3 << 25);           // pad_amount: 4 DWORDs (16B)
    g1[1] = (int)(((unsigned)rem_k & 0xFFFFu) << 16);                       // tensor_dim0 lo
    g1[2] = (int)(((unsigned)rem_k >> 16) | (((unsigned)rem_n & 0xFFFFu) << 16)); // dim0 hi | dim1 lo
    g1[3] = (int)(((unsigned)rem_n >> 16) | (64u << 16));                   // dim1 hi | tile_dim0=64
    g1[4] = 128;                                        // tile_dim1=128, tile_dim2=0
    g1[5] = ld_elems;                                   // tensor_dim0_stride[31:0]
    g1[6] = 0;                                          // stride[47:32] | dim1_stride lo
    g1[7] = 0;
    i32x4 z4 = {0, 0, 0, 0};
#if defined(__clang_major__) && (__clang_major__ >= 23)
    i32x8 z8 = {0, 0, 0, 0, 0, 0, 0, 0};
    __builtin_amdgcn_tensor_load_to_lds(g0, g1, z4, z4, z8, 0);
#else
    __builtin_amdgcn_tensor_load_to_lds(g0, g1, z4, z4, 0);
#endif
}

// ---------------- prep: PE add + f32->bf16 ----------------
__global__ __launch_bounds__(256) void pe_kernel(const float* __restrict__ support,
                                                 const float* __restrict__ queries,
                                                 bf16_t* __restrict__ xpe) {
    int idx = blockIdx.x * 256 + threadIdx.x;              // over 225*8*2048
    if (idx >= N_SAMPLES * SEQ_LEN * D_IN) return;
    int d   = idx & (D_IN - 1);
    int pos = (idx >> 11) & (SEQ_LEN - 1);
    int s   = idx >> 14;
    float x = (s < N_SUPPORT) ? support[idx] : queries[idx - N_SUPPORT * SEQ_LEN * D_IN];
    int   i2  = d >> 1;
    float dv  = __expf((float)(2 * i2) * (-9.210340371976184f / (float)D_IN)); // -ln(1e4)/D
    float ang = (float)pos * dv;
    float pe  = ((d & 1) ? __cosf(ang) : __sinf(ang)) * 0.1f;
    xpe[idx] = (__bf16)(x + pe);
}

// ---------------- prep: tuple gather bases ----------------
__global__ __launch_bounds__(256) void rowbase_kernel(int* __restrict__ rowbase) {
    int r = blockIdx.x * 256 + threadIdx.x;
    if (r >= ROWS_ALL) return;
    int sample = r / T_TUP, t = r % T_TUP;
    int a = 0, b = 1, c = 2, cnt = 0;
    for (int i = 0; i < SEQ_LEN; ++i)
        for (int j = i + 1; j < SEQ_LEN; ++j)
            for (int k = j + 1; k < SEQ_LEN; ++k) {
                if (cnt == t) { a = i; b = j; c = k; }
                cnt++;
            }
    rowbase[r * 3 + 0] = (sample * SEQ_LEN + a) * D_IN;
    rowbase[r * 3 + 1] = (sample * SEQ_LEN + b) * D_IN;
    rowbase[r * 3 + 2] = (sample * SEQ_LEN + c) * D_IN;
}

// ---------------- prep: W [K x N] f32 -> Wt [N x K] bf16 ----------------
__global__ __launch_bounds__(256) void wt_kernel(const float* __restrict__ W,
                                                 bf16_t* __restrict__ Wt) {
    int idx = blockIdx.x * 256 + threadIdx.x;              // over 1152*6144
    if (idx >= D_OUT * K_BIG) return;
    int n = idx / K_BIG, k = idx % K_BIG;
    Wt[idx] = (__bf16)W[(size_t)k * D_OUT + n];
}

// ---------------- generic bf16 WMMA GEMM: C[M,N] = A[M,K] * Bt[N,K]^T ----------------
// A staged by per-lane async b128 copies (ASYNCcnt), B staged by the TDM (TENSORcnt),
// double-buffered LDS so copies for tile k+1 overlap WMMAs on tile k.
template <bool GATHER>
__global__ __launch_bounds__(256) void gemm_bf16_kernel(const bf16_t* __restrict__ A,
                                                        const int* __restrict__ rowbase,
                                                        const bf16_t* __restrict__ Bt,
                                                        float* __restrict__ C,
                                                        int M, int N, int K,
                                                        int lda, int ldb, int ldc,
                                                        const float* __restrict__ bias,
                                                        float scale) {
    __shared__ __align__(16) bf16_t sA[2][128 * 72];   // 64 K padded to 72 (144B rows)
    __shared__ __align__(16) bf16_t sB[2][128 * 72];

    const int tid  = threadIdx.x;
    const int m0   = blockIdx.y * 128;
    const int n0   = blockIdx.x * 128;
    const int wid  = tid >> 5, lane = tid & 31;
    const int wm   = wid >> 2, wn = wid & 3;           // 2 x 4 wave grid
    const int lrow = lane & 15, lhi = lane >> 4;

    // per-thread A staging slots: 4 slots of 16B covering 128 rows x 64 K
    int rb0[4], rb1[4], rb2[4];
    size_t aoff[4];
    int ldsoff[4], seg8[4];
    #pragma unroll
    for (int it = 0; it < 4; ++it) {
        int i = tid + it * 256;
        int row = i >> 3, seg = i & 7;
        int r = min(m0 + row, M - 1);
        ldsoff[it] = row * 72 + seg * 8;
        seg8[it]   = seg * 8;
        if (GATHER) {
            rb0[it] = rowbase[r * 3 + 0];
            rb1[it] = rowbase[r * 3 + 1];
            rb2[it] = rowbase[r * 3 + 2];
            aoff[it] = 0;
        } else {
            aoff[it] = (size_t)r * lda;
        }
    }

    auto stage = [&](int k0, int buf) {
        // A: 4 async b128 per thread
        int j = k0 >> 11;
        int kin = k0 & (D_IN - 1);
        #pragma unroll
        for (int it = 0; it < 4; ++it) {
            const bf16_t* src;
            if (GATHER) {
                int base = (j == 0) ? rb0[it] : ((j == 1) ? rb1[it] : rb2[it]);
                src = A + base + kin + seg8[it];
            } else {
                src = A + aoff[it] + k0 + seg8[it];
            }
            async_b128(&sA[buf][ldsoff[it]], src);
        }
        // B: one TDM descriptor per block (wave 0, lane 0 issues; EXEC-ignored op)
        if (tid == 0)
            tdm_load_tile(&sB[buf][0], Bt + (size_t)n0 * ldb + k0, K - k0, N - n0, ldb);
    };

    f32x8 acc[4][2] = {};
    const int ktiles = K >> 6;

    stage(0, 0);
    for (int kt = 0; kt < ktiles; ++kt) {
        __syncthreads();                                // everyone done reading buf^1
        const int buf = kt & 1;
        const bool more = (kt + 1 < ktiles);
        if (more) stage((kt + 1) << 6, buf ^ 1);
        if (more) wait_async_le4(); else wait_async_0();
        if (wid == 0) {
            if (more) __builtin_amdgcn_s_wait_tensorcnt(1);
            else      __builtin_amdgcn_s_wait_tensorcnt(0);
        }
        __syncthreads();                                // tile kt fully in LDS

        #pragma unroll
        for (int s = 0; s < 2; ++s) {
            Frag a[4], b[2];
            #pragma unroll
            for (int mt = 0; mt < 4; ++mt) {
                int base = (wm * 64 + mt * 16 + lrow) * 72 + s * 32 + lhi * 8;
                a[mt].h[0] = *(const bf16x8*)&sA[buf][base];
                a[mt].h[1] = *(const bf16x8*)&sA[buf][base + 16];
            }
            #pragma unroll
            for (int nt = 0; nt < 2; ++nt) {
                int base = (wn * 32 + nt * 16 + lrow) * 72 + s * 32 + lhi * 8;
                b[nt].h[0] = *(const bf16x8*)&sB[buf][base];
                b[nt].h[1] = *(const bf16x8*)&sB[buf][base + 16];
            }
            #pragma unroll
            for (int mt = 0; mt < 4; ++mt)
                #pragma unroll
                for (int nt = 0; nt < 2; ++nt)
                    acc[mt][nt] = wmma_bf16(a[mt].v, b[nt].v, acc[mt][nt]);
        }
    }

    // ---- epilogue: scale + bias, guarded store ----
    #pragma unroll
    for (int mt = 0; mt < 4; ++mt)
        #pragma unroll
        for (int nt = 0; nt < 2; ++nt)
            #pragma unroll
            for (int e = 0; e < 8; ++e) {
                int m = m0 + wm * 64 + mt * 16 + e + lhi * 8;
                int n = n0 + wn * 32 + nt * 16 + lrow;
                if (m < M && n < N) {
                    float o = acc[mt][nt][e] * scale;
                    if (bias) o += bias[n];
                    C[(size_t)m * ldc + n] = o;
                }
            }
}

// ---------------- LayerNorm row kernel: f32 in -> bf16 out ----------------
__global__ __launch_bounds__(256) void ln_kernel(const float* __restrict__ X,
                                                 const float* __restrict__ w,
                                                 const float* __restrict__ b,
                                                 bf16_t* __restrict__ Y) {
    __shared__ float s1[256], s2[256];
    int row = blockIdx.x, tid = threadIdx.x;
    const float* x = X + (size_t)row * D_OUT;
    float a = 0.f, a2 = 0.f;
    for (int i = tid; i < D_OUT; i += 256) { float v = x[i]; a += v; a2 += v * v; }
    s1[tid] = a; s2[tid] = a2; __syncthreads();
    for (int s = 128; s > 0; s >>= 1) {
        if (tid < s) { s1[tid] += s1[tid + s]; s2[tid] += s2[tid + s]; }
        __syncthreads();
    }
    float mean = s1[0] / (float)D_OUT;
    float var  = s2[0] / (float)D_OUT - mean * mean;
    float inv  = rsqrtf(var + 1e-5f);
    bf16_t* y = Y + (size_t)row * D_OUT;
    for (int i = tid; i < D_OUT; i += 256)
        y[i] = (__bf16)((x[i] - mean) * inv * w[i] + b[i]);
}

// ---------------- sv (f32 rows) -> cvT [D_OUT x CVT_LD] bf16, zero padded ----------------
__global__ __launch_bounds__(256) void cvt_kernel(const float* __restrict__ vall,
                                                  bf16_t* __restrict__ cvT) {
    int idx = blockIdx.x * 256 + threadIdx.x;              // over 1152*1408
    if (idx >= D_OUT * CVT_LD) return;
    int d = idx / CVT_LD, col = idx % CVT_LD;
    cvT[idx] = (col < ROWS_S) ? (__bf16)vall[(size_t)col * D_OUT + d] : (__bf16)0.f;
}

// ---------------- fused softmax + proto GEMM + sq-dist per (query, class) ----------------
__global__ __launch_bounds__(256) void attn_kernel(const float* __restrict__ scores,
                                                   const bf16_t* __restrict__ cvT,
                                                   const float* __restrict__ vall,
                                                   float* __restrict__ out) {
    __shared__ __align__(16) bf16_t sAttn[64 * ATT_LDS];
    __shared__ float red[256];

    const int c = blockIdx.x, q = blockIdx.y;
    const int tid = threadIdx.x;

    // ---- softmax over 280 (shot x tuple) per query-tuple row ----
    int row = tid >> 2, p = tid & 3;                       // 64 rows x 4 partners
    const float* srow = scores + (size_t)(q * T_TUP + row) * ROWS_S + c * (SHOT * T_TUP);
    float mx = -1e30f;
    if (row < T_TUP)
        for (int j = p * 70; j < p * 70 + 70; ++j) mx = fmaxf(mx, srow[j]);
    red[tid] = mx; __syncthreads();
    float m4 = fmaxf(fmaxf(red[row * 4 + 0], red[row * 4 + 1]),
                     fmaxf(red[row * 4 + 2], red[row * 4 + 3]));
    __syncthreads();
    float sum = 0.f;
    if (row < T_TUP) {
        for (int j = p * 70; j < p * 70 + 70; ++j) {
            float e = __expf(srow[j] - m4);
            sum += e;
            sAttn[row * ATT_LDS + j] = (__bf16)e;
        }
    } else {
        for (int j = p * 70; j < p * 70 + 70; ++j) sAttn[row * ATT_LDS + j] = (__bf16)0.f;
    }
    if (p == 3)
        for (int j = 280; j < ATT_K; ++j) sAttn[row * ATT_LDS + j] = (__bf16)0.f;
    red[tid] = sum; __syncthreads();
    float s4 = red[row * 4 + 0] + red[row * 4 + 1] + red[row * 4 + 2] + red[row * 4 + 3];
    float inv = (row < T_TUP) ? 1.f / s4 : 0.f;
    __syncthreads();
    if (row < T_TUP)
        for (int j = p * 70; j < p * 70 + 70; ++j)
            sAttn[row * ATT_LDS + j] = (__bf16)((float)sAttn[row * ATT_LDS + j] * inv);
    __syncthreads();

    // ---- proto GEMM: attn[64 x 288] * cvT(class slice)[288 x 1152], fused diff^2 ----
    const int wid = tid >> 5, lane = tid & 31;
    const int lrow = lane & 15, lhi = lane >> 4;
    const float* qvrow = vall + (size_t)(ROWS_S + q * T_TUP) * D_OUT;
    float ss = 0.f;

    for (int nti = 0; nti < 9; ++nti) {
        int n0 = (nti * 8 + wid) * 16;                     // output dim d tile
        if (nti + 1 < 9) {                                 // hint next B tile into caches
            int nn = ((nti + 1) * 8 + wid) * 16;
            __builtin_prefetch(cvT + (size_t)(nn + lrow) * CVT_LD + c * (SHOT * T_TUP), 0, 3);
        }
        f32x8 acc[4] = {};
        #pragma unroll 3
        for (int ks = 0; ks < 9; ++ks) {
            Frag b;
            const bf16_t* bp = cvT + (size_t)(n0 + lrow) * CVT_LD
                                   + c * (SHOT * T_TUP) + ks * 32 + lhi * 8;
            b.h[0] = *(const bf16x8*)bp;
            b.h[1] = *(const bf16x8*)(bp + 16);
            #pragma unroll
            for (int mt = 0; mt < 4; ++mt) {
                Frag a;
                int ab = (mt * 16 + lrow) * ATT_LDS + ks * 32 + lhi * 8;
                a.h[0] = *(const bf16x8*)&sAttn[ab];
                a.h[1] = *(const bf16x8*)&sAttn[ab + 16];
                acc[mt] = wmma_bf16(a.v, b.v, acc[mt]);
            }
        }
        #pragma unroll
        for (int mt = 0; mt < 4; ++mt)
            #pragma unroll
            for (int e = 0; e < 8; ++e) {
                int m = mt * 16 + e + lhi * 8;
                if (m < T_TUP) {
                    int d = n0 + lrow;
                    float diff = qvrow[(size_t)m * D_OUT + d] - acc[mt][e];
                    ss += diff * diff;
                }
            }
    }

    red[tid] = ss; __syncthreads();
    for (int s = 128; s > 0; s >>= 1) {
        if (tid < s) red[tid] += red[tid + s];
        __syncthreads();
    }
    if (tid == 0) out[q * WAY + c] = -red[0] / (float)T_TUP;
}

// ---------------- host launcher ----------------
extern "C" void kernel_launch(void* const* d_in, const int* in_sizes, int n_in,
                              void* d_out, int out_size, void* d_ws, size_t ws_size,
                              hipStream_t stream) {
    (void)in_sizes; (void)n_in; (void)out_size; (void)ws_size;
    const float* support = (const float*)d_in[0];
    const float* queries = (const float*)d_in[1];
    // d_in[2] = support_labels (sorted class-grouped; unused)
    const float* Wk   = (const float*)d_in[3];
    const float* bk   = (const float*)d_in[4];
    const float* Wv   = (const float*)d_in[5];
    const float* bv   = (const float*)d_in[6];
    const float* ln_w = (const float*)d_in[7];
    const float* ln_b = (const float*)d_in[8];
    float* out = (float*)d_out;

    // workspace carve-up (256B aligned)
    char* ws = (char*)d_ws;
    size_t off = 0;
    auto carve = [&](size_t bytes) {
        char* p = ws + off;
        off = (off + bytes + 255) & ~(size_t)255;
        return p;
    };
    bf16_t* xpe     = (bf16_t*)carve((size_t)N_SAMPLES * SEQ_LEN * D_IN * 2);   // 7.4 MB
    int*    rowbase = (int*)   carve((size_t)ROWS_ALL * 3 * 4);                 // 0.15 MB
    bf16_t* WtK     = (bf16_t*)carve((size_t)D_OUT * K_BIG * 2);                // 14.2 MB
    bf16_t* WtV     = (bf16_t*)carve((size_t)D_OUT * K_BIG * 2);                // 14.2 MB
    float*  kall    = (float*) carve((size_t)ROWS_ALL * D_OUT * 4);             // 58 MB
    float*  vall    = (float*) carve((size_t)ROWS_ALL * D_OUT * 4);             // 58 MB
    bf16_t* kbf     = (bf16_t*)carve((size_t)ROWS_ALL * D_OUT * 2);             // 29 MB
    bf16_t* cvT     = (bf16_t*)carve((size_t)D_OUT * CVT_LD * 2);               // 3.2 MB
    float*  scores  = (float*) carve((size_t)ROWS_Q * ROWS_S * 4);              // 62.7 MB

    // 1. PE add + bf16 convert
    pe_kernel<<<(N_SAMPLES * SEQ_LEN * D_IN + 255) / 256, 256, 0, stream>>>(support, queries, xpe);
    // 2. tuple gather bases
    rowbase_kernel<<<(ROWS_ALL + 255) / 256, 256, 0, stream>>>(rowbase);
    // 3. weight transposes
    wt_kernel<<<(D_OUT * K_BIG + 255) / 256, 256, 0, stream>>>(Wk, WtK);
    wt_kernel<<<(D_OUT * K_BIG + 255) / 256, 256, 0, stream>>>(Wv, WtV);

    // 4/5. big projection GEMMs (gathered A): kall = A*Wk + bk ; vall = A*Wv + bv
    dim3 g_kv((D_OUT + 127) / 128, (ROWS_ALL + 127) / 128);
    gemm_bf16_kernel<true><<<g_kv, 256, 0, stream>>>(xpe, rowbase, WtK, kall,
        ROWS_ALL, D_OUT, K_BIG, 0, K_BIG, D_OUT, bk, 1.0f);
    gemm_bf16_kernel<true><<<g_kv, 256, 0, stream>>>(xpe, rowbase, WtV, vall,
        ROWS_ALL, D_OUT, K_BIG, 0, K_BIG, D_OUT, bv, 1.0f);

    // 6. LayerNorm on keys -> bf16
    ln_kernel<<<ROWS_ALL, 256, 0, stream>>>(kall, ln_w, ln_b, kbf);
    // 7. value transpose for proto GEMM B-operand
    cvt_kernel<<<(D_OUT * CVT_LD + 255) / 256, 256, 0, stream>>>(vall, cvT);

    // 8. attention scores: qk (query rows) x sk^T, scaled by 1/sqrt(D_OUT)
    dim3 g_sc((ROWS_S + 127) / 128, (ROWS_Q + 127) / 128);
    gemm_bf16_kernel<false><<<g_sc, 256, 0, stream>>>(kbf + (size_t)ROWS_S * D_OUT, nullptr,
        kbf, scores, ROWS_Q, ROWS_S, D_OUT, D_OUT, D_OUT, ROWS_S,
        nullptr, 1.0f / sqrtf((float)D_OUT));

    // 9. fused softmax + proto + squared-distance logits
    dim3 g_at(WAY, N_QUERIES);
    attn_kernel<<<g_at, 256, 0, stream>>>(scores, cvT, vall, out);
}
